// FFAttention_72808285601909
// MI455X (gfx1250) — matrix operand order repaired and checked by reading, not compiled, not run on hardware
//
#include <hip/hip_runtime.h>
#include <hip/hip_bf16.h>

// ---------------------------------------------------------------------------
// PVT-style spatially-reduced attention, B=8 C=256 H=W=64, HEADS=8, d=32, R=4
// All GEMMs + attention matmuls on v_wmma_f32_16x16x32_f16 (wave32).
// f16 pack kernels put A/B operands in fragment-contiguous layouts so the
// steady-state GEMM loops are pure global_load_b128 + v_wmma.
// Intermediates (~76 MB) live in d_ws -> fully L2-resident on MI455X (192 MB).
// ---------------------------------------------------------------------------

typedef __attribute__((ext_vector_type(16))) _Float16 v16h;
typedef __attribute__((ext_vector_type(8)))  _Float16 v8h;
typedef __attribute__((ext_vector_type(4)))  _Float16 v4h;
typedef __attribute__((ext_vector_type(8)))  float    v8f;

#define HEADS 8
#define CCH   256
#define DH    32
#define NTOK  4096     // H*W per batch
#define NRED  256      // (H/4)*(W/4) per batch
#define KCONV 4096     // C*R*R
#define HH    64
#define WWID  64

__device__ __forceinline__ v8f wmma_f16(v16h a, v16h b, v8f c) {
  return __builtin_amdgcn_wmma_f32_16x16x32_f16(false, a, false, b, (short)0, c,
                                                false, false);
}
__device__ __forceinline__ v16h cat8(v8h lo, v8h hi) {
  return __builtin_shufflevector(lo, hi, 0, 1, 2, 3, 4, 5, 6, 7,
                                 8, 9, 10, 11, 12, 13, 14, 15);
}
__device__ __forceinline__ v8h ld8(const _Float16* p) { return *(const v8h*)p; }

// Fragment coordinate maps (wave32, ISA 7.12.2):
//  A 16x32 f16 : m = L&15 ; k = ((L>>4)<<3) + (e&7) + ((e>>3)<<4)
//                -> code instead uses k = ((L>>4)<<4) + e: 16 k-contiguous per lane (afrag at 2*kb)
//  B 32x16 f16 : ISA k order as A; code uses k = ((L>>4)<<4) + e (same as A) ; n = L&15  -> 16 k-contiguous per lane
//  C/D 16x16 f32: m = ((L>>4)<<3) + r ; n = L&15

// A fragment from a k-contiguous f16 row
__device__ __forceinline__ v16h afrag(const _Float16* row, int kb) {
  return cat8(ld8(row + 2 * kb), ld8(row + 2 * kb + 8));
}
// B fragment from a k-contiguous f16 row (weights stored [n][k])
__device__ __forceinline__ v16h bfrag(const _Float16* row, int kb) {
  return cat8(ld8(row + kb), ld8(row + kb + 8));
}

// ---------------------------------------------------------------------------
// P0: f32 -> f16 vectorized convert (weights).  n4 = n/4 chunks.
// ---------------------------------------------------------------------------
__global__ __launch_bounds__(256)
void cvt4_f32_f16(const float* __restrict__ s, _Float16* __restrict__ d, int n4) {
  int i = blockIdx.x * 256 + threadIdx.x;
  if (i >= n4) return;
  float4 v = ((const float4*)s)[i];
  v4h o;
  o[0] = (_Float16)v.x; o[1] = (_Float16)v.y;
  o[2] = (_Float16)v.z; o[3] = (_Float16)v.w;
  ((v4h*)d)[i] = o;
}

// ---------------------------------------------------------------------------
// P1: im2col pack for the stride-4 conv.  im2[t_r][k], k = c*16 + i*4 + j.
// ---------------------------------------------------------------------------
__global__ __launch_bounds__(256)
void pack_im2col(const float* __restrict__ x, _Float16* __restrict__ im2) {
  int gid = blockIdx.x * 256 + threadIdx.x;   // 2048 * 1024
  int t  = gid >> 10;
  int kc = gid & 1023;
  int k  = kc * 4;                            // j = 0..3
  int c = k >> 4, i = (k >> 2) & 3;
  int b = t >> 8, th = (t >> 4) & 15, tw = t & 15;
  const float* p = &x[(((b * CCH + c) * HH) + th * 4 + i) * WWID + tw * 4];
  float4 v = *(const float4*)p;
  v4h o;
  o[0] = (_Float16)v.x; o[1] = (_Float16)v.y;
  o[2] = (_Float16)v.z; o[3] = (_Float16)v.w;
  *(v4h*)&im2[t * KCONV + k] = o;
}

// ---------------------------------------------------------------------------
// P2: token-major f16 transpose of x:  xt[b][pix][c] = x[b][c][pix]
// ---------------------------------------------------------------------------
__global__ __launch_bounds__(256)
void pack_xt(const float* __restrict__ x, _Float16* __restrict__ xt) {
  int gid = blockIdx.x * 256 + threadIdx.x;   // 8 * 64 * 4096
  int pix = gid & (NTOK - 1);
  int cc  = ((gid >> 12) & 63) * 4;
  int b   = gid >> 18;
  v4h o;
#pragma unroll
  for (int u = 0; u < 4; ++u)
    o[u] = (_Float16)x[(b * CCH + cc + u) * NTOK + pix];
  *(v4h*)&xt[(b * NTOK + pix) * CCH + cc] = o;
}

// ---------------------------------------------------------------------------
// K1a: conv GEMM.  M = 2048, K = 4096, N = 256 (16x64 tile per wave).
// ---------------------------------------------------------------------------
__global__ __launch_bounds__(32)
void sr_conv_gemm(const _Float16* __restrict__ im2,
                  const _Float16* __restrict__ wsr, float* __restrict__ xr) {
  const int lane = threadIdx.x & 31;
  const int m0 = blockIdx.x * 16;
  const int n0 = blockIdx.y * 64;
  const int l15 = lane & 15;
  const int akb = (lane >> 4) << 3;
  const int bkb = (lane >> 4) << 4;
  const _Float16* arow = im2 + (size_t)(m0 + l15) * KCONV;
  const _Float16* brow0 = wsr + (size_t)(n0 + l15) * KCONV;
  const _Float16* brow1 = brow0 + (size_t)16 * KCONV;
  const _Float16* brow2 = brow0 + (size_t)32 * KCONV;
  const _Float16* brow3 = brow0 + (size_t)48 * KCONV;

  v8f a0 = {}, a1 = {}, a2 = {}, a3 = {};
  for (int k0 = 0; k0 < KCONV; k0 += 32) {
    if (k0 + 32 < KCONV) __builtin_prefetch(arow + k0 + 32 + akb, 0, 0);
    v16h a = afrag(arow + k0, akb);
    a0 = wmma_f16(a, bfrag(brow0 + k0, bkb), a0);
    a1 = wmma_f16(a, bfrag(brow1 + k0, bkb), a1);
    a2 = wmma_f16(a, bfrag(brow2 + k0, bkb), a2);
    a3 = wmma_f16(a, bfrag(brow3 + k0, bkb), a3);
  }
#pragma unroll
  for (int r = 0; r < 8; ++r) {
    int m = ((lane >> 4) << 3) + r;
    float* yr = &xr[(m0 + m) * CCH + n0 + l15];
    yr[0] = a0[r]; yr[16] = a1[r]; yr[32] = a2[r]; yr[48] = a3[r];
  }
}

// ---------------------------------------------------------------------------
// K1b: LayerNorm over channels (biased variance), one wave per token.
// ---------------------------------------------------------------------------
__global__ __launch_bounds__(32)
void layernorm_tok(const float* __restrict__ xr, const float* __restrict__ g,
                   const float* __restrict__ beta, _Float16* __restrict__ xrn) {
  const int t = blockIdx.x;
  const int lane = threadIdx.x & 31;
  float v[8];
  float s = 0.f;
#pragma unroll
  for (int i = 0; i < 8; ++i) { v[i] = xr[t * CCH + lane + 32 * i]; s += v[i]; }
#pragma unroll
  for (int off = 1; off < 32; off <<= 1) s += __shfl_xor(s, off, 32);
  const float mu = s * (1.0f / CCH);
  float sq = 0.f;
#pragma unroll
  for (int i = 0; i < 8; ++i) { float d = v[i] - mu; sq += d * d; }
#pragma unroll
  for (int off = 1; off < 32; off <<= 1) sq += __shfl_xor(sq, off, 32);
  const float rstd = rsqrtf(sq * (1.0f / CCH) + 1e-6f);
#pragma unroll
  for (int i = 0; i < 8; ++i) {
    int c = lane + 32 * i;
    xrn[t * CCH + c] = (_Float16)((v[i] - mu) * rstd * g[c] + beta[c]);
  }
}

// ---------------------------------------------------------------------------
// K2: KV projection.  M = 2048, K = 256, N = 512 (16x64 tile per wave).
//     Also writes V transposed: vt[b][h][dd][j] for the P*V B-fragments.
// ---------------------------------------------------------------------------
__global__ __launch_bounds__(32)
void kv_gemm(const _Float16* __restrict__ xrn, const _Float16* __restrict__ wkv,
             _Float16* __restrict__ kv, _Float16* __restrict__ vt) {
  const int lane = threadIdx.x & 31;
  const int m0 = blockIdx.x * 16;
  const int n0 = blockIdx.y * 64;
  const int l15 = lane & 15;
  const int akb = (lane >> 4) << 3;
  const int bkb = (lane >> 4) << 4;
  const _Float16* arow = xrn + (size_t)(m0 + l15) * CCH;
  const _Float16* brow0 = wkv + (size_t)(n0 + l15) * CCH;

  v8f a0 = {}, a1 = {}, a2 = {}, a3 = {};
  for (int k0 = 0; k0 < CCH; k0 += 32) {
    v16h a = afrag(arow + k0, akb);
    a0 = wmma_f16(a, bfrag(brow0 + k0, bkb), a0);
    a1 = wmma_f16(a, bfrag(brow0 + 16 * CCH + k0, bkb), a1);
    a2 = wmma_f16(a, bfrag(brow0 + 32 * CCH + k0, bkb), a2);
    a3 = wmma_f16(a, bfrag(brow0 + 48 * CCH + k0, bkb), a3);
  }
#pragma unroll
  for (int r = 0; r < 8; ++r) {
    int m = ((lane >> 4) << 3) + r;
    int t = m0 + m;
    _Float16* row = &kv[t * (2 * CCH)];
    float vv[4] = { a0[r], a1[r], a2[r], a3[r] };
#pragma unroll
    for (int nt = 0; nt < 4; ++nt) {
      int col = n0 + nt * 16 + l15;
      _Float16 hv = (_Float16)vv[nt];
      row[col] = hv;
      if (col >= CCH) {  // V half -> also store transposed per-head
        int cv = col - CCH;
        int h = cv >> 5, dd = cv & 31;
        int b = t >> 8, jr = t & (NRED - 1);
        vt[((b * HEADS + h) * DH + dd) * NRED + jr] = hv;
      }
    }
  }
}

// ---------------------------------------------------------------------------
// K3: Q projection (softmax scale folded in).  M = 32768, K = 256, N = 256.
// ---------------------------------------------------------------------------
__global__ __launch_bounds__(32)
void q_gemm(const _Float16* __restrict__ xt, const _Float16* __restrict__ wq,
            _Float16* __restrict__ q) {
  const int lane = threadIdx.x & 31;
  const int m0 = blockIdx.x * 16;
  const int n0 = blockIdx.y * 64;
  const int l15 = lane & 15;
  const int akb = (lane >> 4) << 3;
  const int bkb = (lane >> 4) << 4;
  const float scale = 0.17677669529663687f;  // 32^-0.5
  const _Float16* arow = xt + (size_t)(m0 + l15) * CCH;
  const _Float16* brow0 = wq + (size_t)(n0 + l15) * CCH;

  v8f a0 = {}, a1 = {}, a2 = {}, a3 = {};
  for (int k0 = 0; k0 < CCH; k0 += 32) {
    if (k0 + 32 < CCH) __builtin_prefetch(arow + k0 + 32, 0, 0);
    v16h a = afrag(arow + k0, akb);
    a0 = wmma_f16(a, bfrag(brow0 + k0, bkb), a0);
    a1 = wmma_f16(a, bfrag(brow0 + 16 * CCH + k0, bkb), a1);
    a2 = wmma_f16(a, bfrag(brow0 + 32 * CCH + k0, bkb), a2);
    a3 = wmma_f16(a, bfrag(brow0 + 48 * CCH + k0, bkb), a3);
  }
#pragma unroll
  for (int r = 0; r < 8; ++r) {
    int m = ((lane >> 4) << 3) + r;
    _Float16* row = &q[(m0 + m) * CCH + n0 + l15];
    row[0]  = (_Float16)(a0[r] * scale);
    row[16] = (_Float16)(a1[r] * scale);
    row[32] = (_Float16)(a2[r] * scale);
    row[48] = (_Float16)(a3[r] * scale);
  }
}

// ---------------------------------------------------------------------------
// K4: attention.  One wave = 16 queries x 1 head.  N_keys = 256 per batch.
// ---------------------------------------------------------------------------
__global__ __launch_bounds__(32)
void attention(const _Float16* __restrict__ q, const _Float16* __restrict__ kv,
               const _Float16* __restrict__ vt, _Float16* __restrict__ o) {
  __shared__ _Float16 P[16 * NRED];           // 8 KB
  const int lane = threadIdx.x & 31;
  const int m0   = blockIdx.x * 16;           // query tile over B*NTOK
  const int h    = blockIdx.y;
  const int b    = m0 >> 12;
  const int half = lane >> 4;
  const int l15  = lane & 15;

  // Q A-fragment: K dim == head dim (32) -> exactly one fragment
  const _Float16* qrow = &q[(size_t)(m0 + l15) * CCH + h * DH];
  v16h af = cat8(ld8(qrow + (half << 4)), ld8(qrow + (half << 4) + 8));

  // sim = Q * K^T : B[k=dd][n=j] = K[j][dd] (k-contiguous in kv rows)
  v8f S[16];
#pragma unroll
  for (int jt = 0; jt < 16; ++jt) {
    const _Float16* krow =
        &kv[(size_t)(b * NRED + jt * 16 + l15) * (2 * CCH) + h * DH + (half << 4)];
    v16h bf = cat8(ld8(krow), ld8(krow + 8));
    v8f z = {};
    S[jt] = wmma_f16(af, bf, z);
  }

  // softmax over 256 keys; row m = (half<<3)+r lives in one 16-lane half
  float rsum[8];
#pragma unroll
  for (int r = 0; r < 8; ++r) {
    float mx = -1e30f;
#pragma unroll
    for (int jt = 0; jt < 16; ++jt) mx = fmaxf(mx, S[jt][r]);
    mx = fmaxf(mx, __shfl_xor(mx, 1, 32));
    mx = fmaxf(mx, __shfl_xor(mx, 2, 32));
    mx = fmaxf(mx, __shfl_xor(mx, 4, 32));
    mx = fmaxf(mx, __shfl_xor(mx, 8, 32));
    float sm = 0.f;
#pragma unroll
    for (int jt = 0; jt < 16; ++jt) {
      float p = __expf(S[jt][r] - mx);
      S[jt][r] = p;
      sm += p;
    }
    sm += __shfl_xor(sm, 1, 32);
    sm += __shfl_xor(sm, 2, 32);
    sm += __shfl_xor(sm, 4, 32);
    sm += __shfl_xor(sm, 8, 32);
    rsum[r] = sm;
  }

  // P: C-layout -> row-major LDS [m][j] (f16)
#pragma unroll
  for (int jt = 0; jt < 16; ++jt) {
#pragma unroll
    for (int r = 0; r < 8; ++r) {
      int m = (half << 3) + r;
      P[m * NRED + jt * 16 + l15] = (_Float16)S[jt][r];
    }
  }
  __syncthreads();

  // out = P * V^T-packed ; head dim 32 -> two 16-col accumulators
  const _Float16* v0row = &vt[(size_t)((b * HEADS + h) * DH + l15) * NRED];
  const _Float16* v1row = v0row + 16 * NRED;
  const _Float16* prow = &P[l15 * NRED];
  v8f O0 = {}, O1 = {};
  for (int kc = 0; kc < NRED; kc += 32) {
    v16h ap = cat8(ld8(prow + kc + (half << 4)),
                   ld8(prow + kc + (half << 4) + 8));
    v16h b0 = cat8(ld8(v0row + kc + (half << 4)), ld8(v0row + kc + (half << 4) + 8));
    v16h b1 = cat8(ld8(v1row + kc + (half << 4)), ld8(v1row + kc + (half << 4) + 8));
    O0 = wmma_f16(ap, b0, O0);
    O1 = wmma_f16(ap, b1, O1);
  }

#pragma unroll
  for (int r = 0; r < 8; ++r) {
    int m = (half << 3) + r;
    float inv = 1.0f / rsum[r];
    o[(size_t)(m0 + m) * CCH + h * DH + l15]      = (_Float16)(O0[r] * inv);
    o[(size_t)(m0 + m) * CCH + h * DH + 16 + l15] = (_Float16)(O1[r] * inv);
  }
}

// ---------------------------------------------------------------------------
// K5: output projection + residual.  y = x + gamma * (O @ Wout^T)
//     M = 32768, K = 256, N = 256 (16x64 tile).  Scatter back to NCHW fp32.
// ---------------------------------------------------------------------------
__global__ __launch_bounds__(32)
void out_proj(const _Float16* __restrict__ o, const _Float16* __restrict__ wo,
              const float* __restrict__ x, const float* __restrict__ gamma,
              float* __restrict__ y) {
  const int lane = threadIdx.x & 31;
  const int m0 = blockIdx.x * 16;
  const int n0 = blockIdx.y * 64;
  const int l15 = lane & 15;
  const int akb = (lane >> 4) << 3;
  const int bkb = (lane >> 4) << 4;
  const _Float16* arow = o + (size_t)(m0 + l15) * CCH;
  const _Float16* brow0 = wo + (size_t)(n0 + l15) * CCH;

  v8f a0 = {}, a1 = {}, a2 = {}, a3 = {};
  for (int k0 = 0; k0 < CCH; k0 += 32) {
    v16h a = afrag(arow + k0, akb);
    a0 = wmma_f16(a, bfrag(brow0 + k0, bkb), a0);
    a1 = wmma_f16(a, bfrag(brow0 + 16 * CCH + k0, bkb), a1);
    a2 = wmma_f16(a, bfrag(brow0 + 32 * CCH + k0, bkb), a2);
    a3 = wmma_f16(a, bfrag(brow0 + 48 * CCH + k0, bkb), a3);
  }
  const float gs = gamma[0];
#pragma unroll
  for (int r = 0; r < 8; ++r) {
    int m = ((lane >> 4) << 3) + r;
    int t = m0 + m;
    int bb = t >> 12, pix = t & (NTOK - 1);
    float vv[4] = { a0[r], a1[r], a2[r], a3[r] };
#pragma unroll
    for (int nt = 0; nt < 4; ++nt) {
      int n = n0 + nt * 16 + l15;
      size_t idx = ((size_t)(bb * CCH + n)) * NTOK + pix;
      y[idx] = x[idx] + gs * vv[nt];
    }
  }
}

// ---------------------------------------------------------------------------
extern "C" void kernel_launch(void* const* d_in, const int* in_sizes, int n_in,
                              void* d_out, int out_size, void* d_ws, size_t ws_size,
                              hipStream_t stream) {
  const float* x      = (const float*)d_in[0];
  const float* w_q    = (const float*)d_in[1];
  const float* w_kv   = (const float*)d_in[2];
  const float* w_sr   = (const float*)d_in[3];
  const float* norm_g = (const float*)d_in[4];
  const float* norm_b = (const float*)d_in[5];
  const float* w_out  = (const float*)d_in[6];
  const float* gamma  = (const float*)d_in[7];
  float* y = (float*)d_out;

  // workspace layout (all L2-resident on MI455X: ~76 MB << 192 MB L2)
  char* ws = (char*)d_ws;
  size_t off = 0;
  _Float16* im2  = (_Float16*)(ws + off); off += (size_t)2048 * KCONV * 2;  // 16.8 MB
  _Float16* xt16 = (_Float16*)(ws + off); off += (size_t)32768 * CCH * 2;   // 16.8 MB
  _Float16* wq16 = (_Float16*)(ws + off); off += (size_t)CCH * CCH * 2;
  _Float16* wkv16= (_Float16*)(ws + off); off += (size_t)2 * CCH * CCH * 2;
  _Float16* wsr16= (_Float16*)(ws + off); off += (size_t)CCH * KCONV * 2;   // 2 MB
  _Float16* wo16 = (_Float16*)(ws + off); off += (size_t)CCH * CCH * 2;
  float*    xr   = (float*)(ws + off);    off += (size_t)2048 * CCH * 4;    // 2 MB
  _Float16* xrn  = (_Float16*)(ws + off); off += (size_t)2048 * CCH * 2;
  _Float16* kvb  = (_Float16*)(ws + off); off += (size_t)2048 * 2 * CCH * 2;
  _Float16* vt   = (_Float16*)(ws + off); off += (size_t)8 * HEADS * DH * NRED * 2;
  _Float16* qb   = (_Float16*)(ws + off); off += (size_t)32768 * CCH * 2;   // 16.8 MB
  _Float16* ob   = (_Float16*)(ws + off); off += (size_t)32768 * CCH * 2;   // 16.8 MB
  (void)off; (void)ws_size; (void)in_sizes; (void)n_in; (void)out_size;

  // pack: weights + x into f16 fragment-contiguous layouts
  cvt4_f32_f16 <<<(CCH * CCH / 4 + 255) / 256, 256, 0, stream>>>(w_q,  wq16,  CCH * CCH / 4);
  cvt4_f32_f16 <<<(2 * CCH * CCH / 4 + 255) / 256, 256, 0, stream>>>(w_kv, wkv16, 2 * CCH * CCH / 4);
  cvt4_f32_f16 <<<(CCH * KCONV / 4 + 255) / 256, 256, 0, stream>>>(w_sr, wsr16, CCH * KCONV / 4);
  cvt4_f32_f16 <<<(CCH * CCH / 4 + 255) / 256, 256, 0, stream>>>(w_out, wo16, CCH * CCH / 4);
  pack_im2col  <<<(2048 * 1024) / 256, 256, 0, stream>>>(x, im2);
  pack_xt      <<<(8 * 64 * NTOK) / 256, 256, 0, stream>>>(x, xt16);

  // pipeline
  sr_conv_gemm <<<dim3(128, 4),  32, 0, stream>>>(im2, wsr16, xr);
  layernorm_tok<<<2048,          32, 0, stream>>>(xr, norm_g, norm_b, xrn);
  kv_gemm      <<<dim3(128, 8),  32, 0, stream>>>(xrn, wkv16, kvb, vt);
  q_gemm       <<<dim3(2048, 4), 32, 0, stream>>>(xt16, wq16, qb);
  attention    <<<dim3(2048, 8), 32, 0, stream>>>(qb, kvb, vt, ob);
  out_proj     <<<dim3(2048, 4), 32, 0, stream>>>(ob, wo16, x, gamma, y);
}
